// TRTSelfAttn_83940840833310
// MI455X (gfx1250) — matrix operand order repaired and checked by reading, not compiled, not run on hardware
//
#include <hip/hip_runtime.h>
#include <hip/hip_bf16.h>

// Problem constants (match reference)
#define HIDDEN 1024
#define NHEADS 16
#define DHEAD  64
#define SEQ    2048
#define BATCH  2
#define ROWS   (SEQ * BATCH)   // 4096 GEMM rows (s,b flattened, x is (S,B,H))
#define NOUT   (3 * HIDDEN)    // 3072 GEMM cols
#define NKT    (SEQ / 128)     // key tiles in attention

typedef __attribute__((ext_vector_type(16))) __bf16 v16bf;
typedef __attribute__((ext_vector_type(8)))  __bf16 v8bf;
typedef __attribute__((ext_vector_type(8)))  float  v8f;
typedef __attribute__((ext_vector_type(4)))  unsigned int v4u;
typedef __attribute__((ext_vector_type(8)))  int v8i;
typedef __attribute__((ext_vector_type(4)))  int v4i;

// ---- TDM availability (device pass only; host pass takes manual path) -----
#if defined(__HIP_DEVICE_COMPILE__) && defined(__has_builtin)
#  if __has_builtin(__builtin_amdgcn_tensor_load_to_lds) && \
      __has_builtin(__builtin_amdgcn_s_wait_tensorcnt)
#    define USE_TDM 1
#  else
#    define USE_TDM 0
#  endif
#else
#  define USE_TDM 0
#endif

static __device__ inline v8f wmma_bf16(v16bf a, v16bf b, v8f c) {
  // D = A(16x32 bf16) * B(32x16 bf16) + C(16x16 f32)
  return __builtin_amdgcn_wmma_f32_16x16x32_bf16(false, a, false, b, (short)0, c,
                                                 false, false);
}

// A fragment: 16(M) x 32(K), row-major tile in LDS with `stride` halfwords/row.
// ISA 7.12.2: m = lane&15; lanes 0-15 hold K kb..kb+7 & kb+16..kb+23 (kb=0),
// lanes 16-31 the same with kb=8.  Two contiguous 16B ds_load_b128 per lane.
static __device__ inline v16bf load_frag_a(const __bf16* base, int stride, int lane) {
  int m  = lane & 15;
  int kb = (lane >> 4) << 3;
  const __bf16* p = base + m * stride + kb;
  v8bf lo = *(const v8bf*)(p);
  v8bf hi = *(const v8bf*)(p + 16);
  v16bf r;
#pragma unroll
  for (int i = 0; i < 8; ++i) { r[i] = lo[i]; r[i + 8] = hi[i]; }
  return r;
}

// B fragment: 32(K) x 16(N); LDS tile stored row-major as [N][K].
// n = lane&15; lanes 0-15 hold K 0..7 & 16..23, lanes 16-31 hold K 8..15 & 24..31.
static __device__ inline v16bf load_frag_b(const __bf16* base, int stride, int lane) {
  int n  = lane & 15;
  int ko = (lane >> 4) << 3;
  const __bf16* p = base + n * stride + ko;
  v8bf lo = *(const v8bf*)(p);
  v8bf hi = *(const v8bf*)(p + 16);
  v16bf r;
#pragma unroll
  for (int i = 0; i < 8; ++i) { r[i] = lo[i]; r[i + 8] = hi[i]; }
  return r;
}

#if USE_TDM
// Issue a 2D TDM tile load (bf16 elements), global -> LDS, with LDS row
// padding.  D# packing per CDNA5 ISA:
//   group0: count=1 | lds_addr | global_addr[56:0] | type=2
//   group1: data_size=1(2B), pad_enable, pad_interval/amount codes,
//           tensor_dim0/1 = tile_d0/d1 (no OOB), tile_dim0/1, dim0_stride.
// pad codes: interval -> (1<<code)*8 bytes, amount -> (code+1)*4 bytes.
static __device__ inline void tdm_load_tile_2d(
    const __bf16* gsrc, unsigned int lds_byte_addr,
    unsigned int tile_d0, unsigned int tile_d1,
    unsigned long long row_stride_elems,
    unsigned int pad_interval_code, unsigned int pad_amount_code) {
  unsigned long long ga = (unsigned long long)(const void*)gsrc;
  v4u g0;
  g0[0] = 1u;                                            // count=1, user D#
  g0[1] = lds_byte_addr;                                 // bits 63:32
  g0[2] = (unsigned int)ga;                              // global_addr[31:0]
  g0[3] = (unsigned int)((ga >> 32) & 0x01FFFFFFu) | (2u << 30); // [56:32]|type=2
  v8i g1;
  g1[0] = (int)((1u << 16)                     // data_size = 2 bytes
                | (1u << 20)                   // pad_enable
                | (pad_interval_code << 22)
                | (pad_amount_code << 25));    // workgroup_mask = 0
  g1[1] = (int)((tile_d0 & 0xFFFFu) << 16);    // tensor_dim0[15:0]
  g1[2] = (int)((tile_d1 & 0xFFFFu) << 16);    // tensor_dim0[31:16]=0 | dim1[15:0]
  g1[3] = (int)((tile_d0 & 0xFFFFu) << 16);    // tensor_dim1[31:16]=0 | tile_dim0
  g1[4] = (int)(tile_d1 & 0xFFFFu);            // tile_dim1 | tile_dim2=0
  g1[5] = (int)(unsigned int)(row_stride_elems & 0xFFFFFFFFu);  // dim0_stride lo
  g1[6] = (int)(unsigned int)((row_stride_elems >> 32) & 0xFFFFu); // stride hi
  g1[7] = 0;                                   // dim1_stride = 0 (2D tile)
  v4i z4 = (v4i){0, 0, 0, 0};
#if __clang_major__ >= 23
  v8i z8 = (v8i){0, 0, 0, 0, 0, 0, 0, 0};
  __builtin_amdgcn_tensor_load_to_lds(g0, g1, z4, z4, z8, 0);
#else
  __builtin_amdgcn_tensor_load_to_lds(g0, g1, z4, z4, 0);
#endif
}
#endif  // USE_TDM

// ---------------------------------------------------------------- kernel 1 --
__global__ void cvt_bf16_kernel(const float* __restrict__ x,
                                const float* __restrict__ W,
                                __bf16* __restrict__ xb,
                                __bf16* __restrict__ Wb) {
  int i = blockIdx.x * blockDim.x + threadIdx.x;
  if (i < ROWS * HIDDEN) xb[i] = (__bf16)x[i];
  if (i < NOUT * HIDDEN) Wb[i] = (__bf16)W[i];
}

// ---------------------------------------------------------------- kernel 2 --
// qkv[r, o] = sum_h x[r,h] * W[o,h] + bias[o]; scatter to q/k/vT head-major.
// TDM double-buffered: wave 0 DMAs tile k+1 while all waves WMMA tile k.
__global__ __launch_bounds__(256) void qkv_gemm_kernel(
    const __bf16* __restrict__ xb, const __bf16* __restrict__ Wb,
    const float* __restrict__ bias,
    __bf16* __restrict__ qb, __bf16* __restrict__ kb, __bf16* __restrict__ vtb) {
  __shared__ __bf16 As[2][128 * 72];  // 128 rows x 64 K (128B rows + 16B pad)
  __shared__ __bf16 Bs[2][128 * 72];

  const int tid   = threadIdx.x;
  const int lane  = tid & 31;
  const int w     = tid >> 5;   // 8 waves
  const int waveM = w & 3;      // 4 waves along M  -> 32 rows each
  const int waveN = w >> 2;     // 2 waves along N  -> 64 cols each
  const int mbase = blockIdx.y * 128;
  const int nbase = blockIdx.x * 128;

  v8f acc[2][4];
#pragma unroll
  for (int mt = 0; mt < 2; ++mt)
#pragma unroll
    for (int nt = 0; nt < 4; ++nt)
      acc[mt][nt] = (v8f){0.f, 0.f, 0.f, 0.f, 0.f, 0.f, 0.f, 0.f};

#if USE_TDM
  if (w == 0) {  // prime the pipeline: tile k0=0 into buffer 0
    tdm_load_tile_2d(xb + (size_t)mbase * HIDDEN,
                     (unsigned int)(size_t)&As[0][0], 64u, 128u, HIDDEN, 4u, 3u);
    tdm_load_tile_2d(Wb + (size_t)nbase * HIDDEN,
                     (unsigned int)(size_t)&Bs[0][0], 64u, 128u, HIDDEN, 4u, 3u);
  }
#endif

  for (int k0 = 0; k0 < HIDDEN; k0 += 64) {
    const int buf = (k0 >> 6) & 1;
#if USE_TDM
    if (w == 0) __builtin_amdgcn_s_wait_tensorcnt(0);  // tile k0 landed
    __syncthreads();                                   // publish tile k0
    if (w == 0 && k0 + 64 < HIDDEN) {                  // DMA k0+64 under WMMA
      tdm_load_tile_2d(xb + (size_t)mbase * HIDDEN + k0 + 64,
                       (unsigned int)(size_t)&As[buf ^ 1][0], 64u, 128u,
                       HIDDEN, 4u, 3u);
      tdm_load_tile_2d(Wb + (size_t)nbase * HIDDEN + k0 + 64,
                       (unsigned int)(size_t)&Bs[buf ^ 1][0], 64u, 128u,
                       HIDDEN, 4u, 3u);
    }
#else
    __syncthreads();
#pragma unroll
    for (int i = 0; i < 4; ++i) {  // 1024 x 16B chunks / 256 threads
      int c  = tid + i * 256;
      int rl = c >> 3;
      int cq = (c & 7) * 8;
      *(v8bf*)(As[buf] + rl * 72 + cq) =
          *(const v8bf*)(xb + (size_t)(mbase + rl) * HIDDEN + k0 + cq);
      *(v8bf*)(Bs[buf] + rl * 72 + cq) =
          *(const v8bf*)(Wb + (size_t)(nbase + rl) * HIDDEN + k0 + cq);
    }
    __syncthreads();
#endif

#pragma unroll
    for (int kk = 0; kk < 2; ++kk) {
      v16bf af[2], bfr[4];
#pragma unroll
      for (int mt = 0; mt < 2; ++mt)
        af[mt] = load_frag_a(As[buf] + (waveM * 32 + mt * 16) * 72 + kk * 32, 72, lane);
#pragma unroll
      for (int nt = 0; nt < 4; ++nt)
        bfr[nt] = load_frag_b(Bs[buf] + (waveN * 64 + nt * 16) * 72 + kk * 32, 72, lane);
#pragma unroll
      for (int mt = 0; mt < 2; ++mt)
#pragma unroll
        for (int nt = 0; nt < 4; ++nt)
          acc[mt][nt] = wmma_bf16(af[mt], bfr[nt], acc[mt][nt]);
    }
  }

  // Epilogue: +bias, split o = head*192 + which*64 + d, store bf16.
  // head/which derived from the lane-invariant tile base -> one cheap /3
  // per 16x16 tile instead of per-element magic-division chains.
  const int lh   = lane >> 4;
  const int ncol = lane & 15;
#pragma unroll
  for (int mt = 0; mt < 2; ++mt)
#pragma unroll
    for (int nt = 0; nt < 4; ++nt) {
      const int Obase = nbase + waveN * 64 + nt * 16;  // multiple of 16
      const int o64   = Obase >> 6;                    // head*3 + which
      const int head  = o64 / 3;
      const int which = o64 - head * 3;
      const int d     = (Obase & 63) + ncol;
      const float bv  = bias[Obase + ncol];
#pragma unroll
      for (int r = 0; r < 8; ++r) {
        int R  = mbase + waveM * 32 + mt * 16 + r + 8 * lh;
        int s  = R >> 1;            // x is (S, B, H), B == 2
        int bb = R & 1;
        int hb = head * 2 + bb;
        __bf16 val = (__bf16)(acc[mt][nt][r] + bv);
        if (which == 0)      qb[((size_t)hb * SEQ + s) * DHEAD + d] = val;
        else if (which == 1) kb[((size_t)hb * SEQ + s) * DHEAD + d] = val;
        else                 vtb[((size_t)hb * DHEAD + d) * SEQ + s] = val; // vT
      }
    }
}

// ---------------------------------------------------------------- kernel 3 --
// Flash attention: block = (head,batch) x 128-query tile; 8 waves x 16 rows.
// K/V tiles double-buffered through the TDM like the GEMM.
__global__ __launch_bounds__(256) void attn_kernel(
    const __bf16* __restrict__ qb, const __bf16* __restrict__ kbuf,
    const __bf16* __restrict__ vtb, float* __restrict__ out) {
  __shared__ __bf16 Qs[128 * 72];      // 128 q x 64 d (128B rows + 16B pad)
  __shared__ __bf16 Ks[2][128 * 72];   // 128 k x 64 d, double buffered
  __shared__ __bf16 Vs[2][64 * 136];   // 64 d x 128 k (vT; 256B+16B rows)
  __shared__ __bf16 Ps[128 * 136];     // probs, per-wave-private 16-row stripes

  const int tid  = threadIdx.x;
  const int lane = tid & 31;
  const int w    = tid >> 5;
  const int lh   = lane >> 4;
  const int ncol = lane & 15;
  const int hb   = blockIdx.y;      // head*2 + batch
  const int head = hb >> 1;
  const int bb   = hb & 1;
  const int q0   = blockIdx.x * 128;
  const float scale = 0.125f;       // 1/sqrt(64)

  // ---- Prologue: Q tile + first K/V tiles.  TDM completes in-order per
  // wave, so tensorcnt<=2 after 3 issues guarantees the Q load finished. ----
#if USE_TDM
  if (w == 0) {
    tdm_load_tile_2d(qb + ((size_t)hb * SEQ + q0) * DHEAD,
                     (unsigned int)(size_t)Qs, 64u, 128u, DHEAD, 4u, 3u);
    tdm_load_tile_2d(kbuf + (size_t)hb * SEQ * DHEAD,
                     (unsigned int)(size_t)&Ks[0][0], 64u, 128u, DHEAD, 4u, 3u);
    tdm_load_tile_2d(vtb + (size_t)hb * DHEAD * SEQ,
                     (unsigned int)(size_t)&Vs[0][0], 128u, 64u, SEQ, 5u, 3u);
    __builtin_amdgcn_s_wait_tensorcnt(2);  // Q resident
  }
#else
#pragma unroll
  for (int i = 0; i < 4; ++i) {
    int c  = tid + i * 256;
    int rl = c >> 3;
    int cq = (c & 7) * 8;
    *(v8bf*)(Qs + rl * 72 + cq) =
        *(const v8bf*)(qb + ((size_t)hb * SEQ + q0 + rl) * DHEAD + cq);
  }
#endif
  __syncthreads();
  v16bf qf[2];
  qf[0] = load_frag_a(Qs + (w * 16) * 72 + 0, 72, lane);
  qf[1] = load_frag_a(Qs + (w * 16) * 72 + 32, 72, lane);

  float row_m[8], row_l[8];
  v8f Oacc[4];
#pragma unroll
  for (int r = 0; r < 8; ++r) { row_m[r] = -1e30f; row_l[r] = 0.f; }
#pragma unroll
  for (int dt = 0; dt < 4; ++dt)
    Oacc[dt] = (v8f){0.f, 0.f, 0.f, 0.f, 0.f, 0.f, 0.f, 0.f};

  for (int kt = 0; kt < NKT; ++kt) {
    const int buf = kt & 1;
#if USE_TDM
    if (w == 0) __builtin_amdgcn_s_wait_tensorcnt(0);  // K/V tile kt landed
    __syncthreads();                                   // publish tile kt
    if (w == 0 && kt + 1 < NKT) {                      // DMA kt+1 under WMMA
      tdm_load_tile_2d(kbuf + ((size_t)hb * SEQ + (kt + 1) * 128) * DHEAD,
                       (unsigned int)(size_t)&Ks[buf ^ 1][0], 64u, 128u,
                       DHEAD, 4u, 3u);
      tdm_load_tile_2d(vtb + (size_t)hb * DHEAD * SEQ + (kt + 1) * 128,
                       (unsigned int)(size_t)&Vs[buf ^ 1][0], 128u, 64u,
                       SEQ, 5u, 3u);
    }
#else
    __syncthreads();
#pragma unroll
    for (int i = 0; i < 4; ++i) {
      int c  = tid + i * 256;
      int rl = c >> 3;
      int cq = (c & 7) * 8;
      *(v8bf*)(Ks[buf] + rl * 72 + cq) =
          *(const v8bf*)(kbuf + ((size_t)hb * SEQ + kt * 128 + rl) * DHEAD + cq);
      int rv = c >> 4;
      int cv = (c & 15) * 8;
      *(v8bf*)(Vs[buf] + rv * 136 + cv) =
          *(const v8bf*)(vtb + ((size_t)hb * DHEAD + rv) * SEQ + kt * 128 + cv);
    }
    if (kt + 1 < NKT) {
      __builtin_prefetch(kbuf + ((size_t)hb * SEQ + (kt + 1) * 128 + (tid >> 1)) * DHEAD, 0, 1);
      __builtin_prefetch(vtb + ((size_t)hb * DHEAD + (tid >> 2)) * SEQ + (kt + 1) * 128, 0, 1);
    }
    __syncthreads();
#endif

    // S = scale * Q K^T   (16 q-rows x 128 keys per wave)
    v8f S[8];
#pragma unroll
    for (int nt = 0; nt < 8; ++nt) {
      S[nt] = (v8f){0.f, 0.f, 0.f, 0.f, 0.f, 0.f, 0.f, 0.f};
      v16bf b0 = load_frag_b(Ks[buf] + (nt * 16) * 72 + 0, 72, lane);
      S[nt] = wmma_bf16(qf[0], b0, S[nt]);
      v16bf b1 = load_frag_b(Ks[buf] + (nt * 16) * 72 + 32, 72, lane);
      S[nt] = wmma_bf16(qf[1], b1, S[nt]);
    }

    // Online softmax. C-layout: row = r + 8*lh duplicated across 16 lanes.
    float mx[8];
#pragma unroll
    for (int r = 0; r < 8; ++r) mx[r] = -1e30f;
#pragma unroll
    for (int nt = 0; nt < 8; ++nt)
#pragma unroll
      for (int r = 0; r < 8; ++r) {
        float sv = S[nt][r] * scale;
        S[nt][r] = sv;
        mx[r] = fmaxf(mx[r], sv);
      }
#pragma unroll
    for (int r = 0; r < 8; ++r) {
      mx[r] = fmaxf(mx[r], __shfl_xor(mx[r], 1, 32));
      mx[r] = fmaxf(mx[r], __shfl_xor(mx[r], 2, 32));
      mx[r] = fmaxf(mx[r], __shfl_xor(mx[r], 4, 32));
      mx[r] = fmaxf(mx[r], __shfl_xor(mx[r], 8, 32));
    }
    float alpha[8];
#pragma unroll
    for (int r = 0; r < 8; ++r) {
      float mn = fmaxf(row_m[r], mx[r]);
      alpha[r] = __expf(row_m[r] - mn);
      row_m[r] = mn;
    }
    float rs[8];
#pragma unroll
    for (int r = 0; r < 8; ++r) rs[r] = 0.f;
#pragma unroll
    for (int nt = 0; nt < 8; ++nt)
#pragma unroll
      for (int r = 0; r < 8; ++r) {
        float p = __expf(S[nt][r] - row_m[r]);
        S[nt][r] = p;
        rs[r] += p;
      }
#pragma unroll
    for (int r = 0; r < 8; ++r) {
      rs[r] += __shfl_xor(rs[r], 1, 32);
      rs[r] += __shfl_xor(rs[r], 2, 32);
      rs[r] += __shfl_xor(rs[r], 4, 32);
      rs[r] += __shfl_xor(rs[r], 8, 32);
      row_l[r] = row_l[r] * alpha[r] + rs[r];
    }
#pragma unroll
    for (int dt = 0; dt < 4; ++dt)
#pragma unroll
      for (int r = 0; r < 8; ++r) Oacc[dt][r] *= alpha[r];

    // C-layout -> A-fragment layout via per-wave-private LDS stripe (rows
    // w*16..w*16+15): same-wave DS ops are in-order, no barrier needed.
#pragma unroll
    for (int nt = 0; nt < 8; ++nt)
#pragma unroll
      for (int r = 0; r < 8; ++r) {
        int m = w * 16 + r + 8 * lh;
        Ps[m * 136 + nt * 16 + ncol] = (__bf16)S[nt][r];
      }

    // O += P * V^T
    v16bf pf[4];
#pragma unroll
    for (int kk = 0; kk < 4; ++kk)
      pf[kk] = load_frag_a(Ps + (w * 16) * 136 + kk * 32, 136, lane);
#pragma unroll
    for (int dt = 0; dt < 4; ++dt)
#pragma unroll
      for (int kk = 0; kk < 4; ++kk) {
        v16bf vf = load_frag_b(Vs[buf] + (dt * 16) * 136 + kk * 32, 136, lane);
        Oacc[dt] = wmma_bf16(pf[kk], vf, Oacc[dt]);
      }
  }

  // Normalize and write out[s][b][head*64 + d] in fp32.
  float inv[8];
#pragma unroll
  for (int r = 0; r < 8; ++r) inv[r] = 1.0f / row_l[r];
#pragma unroll
  for (int dt = 0; dt < 4; ++dt)
#pragma unroll
    for (int r = 0; r < 8; ++r) {
      int m = w * 16 + r + 8 * lh;
      int s = q0 + m;
      int d = dt * 16 + ncol;
      out[((size_t)s * BATCH + bb) * HIDDEN + head * DHEAD + d] = Oacc[dt][r] * inv[r];
    }
}

// --------------------------------------------------------------- launcher --
extern "C" void kernel_launch(void* const* d_in, const int* in_sizes, int n_in,
                              void* d_out, int out_size, void* d_ws, size_t ws_size,
                              hipStream_t stream) {
  (void)in_sizes; (void)n_in; (void)out_size; (void)ws_size;
  const float* x    = (const float*)d_in[0];
  const float* W    = (const float*)d_in[1];
  const float* bias = (const float*)d_in[2];
  float* out = (float*)d_out;

  // Workspace layout (bytes): xb 8M | Wb 6M | q 8M | k 8M | vT 8M  (~38 MB)
  char* ws = (char*)d_ws;
  const size_t SZ_XB = (size_t)ROWS * HIDDEN * sizeof(__bf16);
  const size_t SZ_WB = (size_t)NOUT * HIDDEN * sizeof(__bf16);
  const size_t N_HBSD = (size_t)NHEADS * BATCH * SEQ * DHEAD;
  __bf16* xb  = (__bf16*)(ws);
  __bf16* Wb  = (__bf16*)(ws + SZ_XB);
  __bf16* qb  = (__bf16*)(ws + SZ_XB + SZ_WB);
  __bf16* kb  = qb + N_HBSD;
  __bf16* vtb = kb + N_HBSD;

  cvt_bf16_kernel<<<(ROWS * HIDDEN + 255) / 256, 256, 0, stream>>>(x, W, xb, Wb);
  qkv_gemm_kernel<<<dim3(NOUT / 128, ROWS / 128), 256, 0, stream>>>(xb, Wb, bias,
                                                                    qb, kb, vtb);
  attn_kernel<<<dim3(SEQ / 128, NHEADS * BATCH), 256, 0, stream>>>(qb, kb, vtb, out);
}
